// LSTMAutoencoder_21320217657468
// MI455X (gfx1250) — matrix operand order repaired and checked by reading, not compiled, not run on hardware
//
#include <hip/hip_runtime.h>
#include <hip/hip_bf16.h>

typedef __attribute__((ext_vector_type(16))) _Float16 v16h;
typedef __attribute__((ext_vector_type(8)))  _Float16 v8h;
typedef __attribute__((ext_vector_type(8)))  float    v8f;

#define B_TOT   8192
#define T_SEQ   256
#define DIN     17
#define HENC    32
#define LATD    8
#define HDEC    17
#define NTE     8        // encoder N tiles (4*32/16)
#define NTD     8        // decoder N tiles: each 17-wide gate padded to 32 -> 128/16
#define NWAVE   2
#define RPW     16       // batch rows per wave
#define RPB     32       // batch rows per block
#define NTHR    64

// gfx1250 has native V_TANH_F32 (single TRANS op); fall back to exp form.
__device__ __forceinline__ float hw_tanh(float x) {
#if __has_builtin(__builtin_amdgcn_tanhf)
    return __builtin_amdgcn_tanhf(x);
#else
    return 2.0f * (1.0f / (1.0f + __expf(-2.0f * x))) - 1.0f;
#endif
}
__device__ __forceinline__ float hw_sigmoid(float x) {
    return 0.5f + 0.5f * hw_tanh(0.5f * x);
}

__device__ __forceinline__ v16h zero_v16h() {
    v16h z;
#pragma unroll
    for (int j = 0; j < 16; ++j) z[j] = (_Float16)0.0f;
    return z;
}

// Wave-local LDS producer->consumer sync: all per-step tiles are wave-private,
// so a full workgroup s_barrier is unnecessary. DS ops of a wave complete (and
// are bank-visible to all lanes) once DScnt reaches 0.
__device__ __forceinline__ void wave_sync() {
    __builtin_amdgcn_wave_barrier();
    asm volatile("s_wait_dscnt 0x0" ::: "memory");
    __builtin_amdgcn_wave_barrier();
}

// Gather a 16x32 f16 A-matrix fragment from row-major LDS [16][32].
// lanes 0-15 -> M=lane, halves 0-7 = K0..7,  halves 8-15 = K16..23;
// lanes 16-31 -> M=lane-16, halves 0-7 = K8..15, halves 8-15 = K24..31.
__device__ __forceinline__ v16h load_a_frag(const _Float16* base, int lane) {
    const int m  = lane & 15;
    const int k0 = (lane < 16) ? 0 : 8;
    const _Float16* p = base + m * 32;
    v8h lo = *(const v8h*)(p + k0);
    v8h hi = *(const v8h*)(p + 16 + k0);
    v16h r;
#pragma unroll
    for (int j = 0; j < 8; ++j) { r[j] = lo[j]; r[8 + j] = hi[j]; }
    return r;
}

// Build the x_t A-fragment DIRECTLY from global memory (no LDS staging):
// lane's M-row is contiguous in memory, K-halves are consecutive floats.
// K=17 carries a constant 1.0 so the K=17 row of the Wih B-tiles (which holds
// the gate bias) is added by the WMMA itself; K=18..31 are zero.
__device__ __forceinline__ v16h load_xfrag(const float* xrow, int t, int lane) {
    const int k0 = (lane < 16) ? 0 : 8;
    const float* p = xrow + t * DIN + k0;
    v16h xf = zero_v16h();
#pragma unroll
    for (int j = 0; j < 8; ++j) xf[j] = (_Float16)p[j];
    if (lane < 16) {
        xf[8] = (_Float16)xrow[t * DIN + 16];   // K=16 (last real input)
        xf[9] = (_Float16)1.0f;                 // K=17 -> bias row
    }
    return xf;
}

__global__ __launch_bounds__(NTHR)
void lstm_ae_kernel(const float* __restrict__ x,
                    const float* __restrict__ Wih_e, const float* __restrict__ Whh_e,
                    const float* __restrict__ bih_e, const float* __restrict__ bhh_e,
                    const float* __restrict__ W_encfc, const float* __restrict__ b_encfc,
                    const float* __restrict__ W_decfc, const float* __restrict__ b_decfc,
                    const float* __restrict__ Wih_d, const float* __restrict__ Whh_d,
                    const float* __restrict__ bih_d, const float* __restrict__ bhh_d,
                    float* __restrict__ out, float* __restrict__ z_out)
{
    // Weights pre-swizzled into WMMA B-fragment layout: [tile][lane][16 halves],
    // element (K = j<8 ? (lane<16?0:8)+j : (lane<16?16:24)+j-8, N = tile*16 + (lane&15)).
    // Decoder gate columns are padded 17 -> 32 so tile 2g/2g+1 == gate g lo/hi,
    // letting the cell update run entirely in registers (no gate LDS round-trip).
    __shared__ __align__(16) _Float16 sWihE[NTE][32][16];
    __shared__ __align__(16) _Float16 sWhhE[NTE][32][16];
    __shared__ __align__(16) _Float16 sWihD[NTD][32][16];
    __shared__ __align__(16) _Float16 sWhhD[NTD][32][16];
    __shared__ float sBiasD[NTD * 16];
    __shared__ float sWenc[LATD][HENC];
    __shared__ float sBenc[LATD];
    __shared__ float sWdec[HENC][LATD];
    __shared__ float sBdec[HENC];
    __shared__ __align__(16) _Float16 sX[NWAVE][RPW][32];  // dec_in (constant decoder input)
    __shared__ __align__(16) _Float16 sH[NWAVE][RPW][32];  // hidden state, K-padded

    const int tid   = threadIdx.x;
    const int w     = tid >> 5;
    const int lane  = tid & 31;
    const int bbase = blockIdx.x * RPB + w * RPW;
    const int nn    = lane & 15;              // C/D fragment column
    const int mb    = (lane < 16) ? 0 : 8;    // C/D fragment row base

    // ---------------- init: stage + swizzle all weights into LDS ----------------
    for (int idx = tid; idx < NTE * 32 * 16; idx += NTHR) {
        int jt = idx >> 9, rem = idx & 511, ln = rem >> 4, j = rem & 15;
        int col = jt * 16 + (ln & 15);
        int k = (j < 8) ? (((ln < 16) ? 0 : 8) + j) : (((ln < 16) ? 16 : 24) + (j - 8));
        // encoder: gate columns are the natural 0..127; K=17 row carries the bias
        float wv;
        if (k < DIN)       wv = Wih_e[col * DIN + k];
        else if (k == DIN) wv = bih_e[col] + bhh_e[col];
        else               wv = 0.0f;
        sWihE[jt][ln][j] = (_Float16)wv;
        sWhhE[jt][ln][j] = (_Float16)Whh_e[col * HENC + k];
        // decoder: column col -> gate g = col/32, hid = col%32 (valid hid < 17)
        int g = col >> 5, hid = col & 31;
        sWihD[jt][ln][j] = (_Float16)((hid < HDEC) ? Wih_d[(g * HDEC + hid) * HENC + k] : 0.0f);
        sWhhD[jt][ln][j] = (_Float16)((hid < HDEC && k < HDEC)
                                          ? Whh_d[(g * HDEC + hid) * HDEC + k] : 0.0f);
    }
    for (int c = tid; c < NTD * 16; c += NTHR) {
        int g = c >> 5, hid = c & 31;
        sBiasD[c] = (hid < HDEC) ? (bih_d[g * HDEC + hid] + bhh_d[g * HDEC + hid]) : 0.0f;
    }
    for (int i = tid; i < LATD * HENC; i += NTHR) sWenc[i / HENC][i % HENC] = W_encfc[i];
    for (int i = tid; i < LATD;        i += NTHR) sBenc[i] = b_encfc[i];
    for (int i = tid; i < HENC * LATD; i += NTHR) sWdec[i / LATD][i % LATD] = W_decfc[i];
    for (int i = tid; i < HENC;        i += NTHR) sBdec[i] = b_decfc[i];
    for (int i = tid; i < NWAVE * RPW * 32; i += NTHR) {
        ((_Float16*)sH)[i] = (_Float16)0.0f;
        ((_Float16*)sX)[i] = (_Float16)0.0f;
    }
    __syncthreads();

    // recurrent weights (critical path) resident in VGPRs for the whole loop
    v16h whhE[NTE];
#pragma unroll
    for (int jt = 0; jt < NTE; ++jt) whhE[jt] = *(const v16h*)&sWhhE[jt][lane][0];

    const float* xrow = x + (size_t)(bbase + (lane & 15)) * T_SEQ * DIN;

    float cL[8], cH[8];
#pragma unroll
    for (int r = 0; r < 8; ++r) { cL[r] = 0.0f; cH[r] = 0.0f; }

    // ================================ encoder ================================
    // software-pipelined x fragment: t+1 loads issue before t's WMMAs retire
    v16h xf = load_xfrag(xrow, 0, lane);
    for (int t = 0; t < T_SEQ; ++t) {
        v16h xn = (t + 1 < T_SEQ) ? load_xfrag(xrow, t + 1, lane) : zero_v16h();
        if (t + 2 < T_SEQ) __builtin_prefetch(xrow + (t + 2) * DIN, 0, 0);

        v16h hfrag = load_a_frag(&sH[w][0][0], lane);
        v8f acc[NTE];
#pragma unroll
        for (int jt = 0; jt < NTE; ++jt) {
            v8f a = {};
            v16h bw = *(const v16h*)&sWihE[jt][lane][0];
            a = __builtin_amdgcn_wmma_f32_16x16x32_f16(false, xf, false, bw,
                                                       (short)0, a, false, false);
            acc[jt] = __builtin_amdgcn_wmma_f32_16x16x32_f16(false, hfrag, false, whhE[jt],
                                                             (short)0, a, false, false);
        }
        // LSTM cell entirely in registers: tile 2g = gate g @ hid=nn, 2g+1 @ hid=nn+16
        // (bias already added by the WMMA via the K=17 row)
#pragma unroll
        for (int r = 0; r < 8; ++r) {
            float gi = hw_sigmoid(acc[0][r]);
            float gf = hw_sigmoid(acc[2][r]);
            float gg = hw_tanh   (acc[4][r]);
            float go = hw_sigmoid(acc[6][r]);
            float c  = gf * cL[r] + gi * gg;  cL[r] = c;
            sH[w][mb + r][nn] = (_Float16)(go * hw_tanh(c));

            gi = hw_sigmoid(acc[1][r]);
            gf = hw_sigmoid(acc[3][r]);
            gg = hw_tanh   (acc[5][r]);
            go = hw_sigmoid(acc[7][r]);
            c  = gf * cH[r] + gi * gg;  cH[r] = c;
            sH[w][mb + r][16 + nn] = (_Float16)(go * hw_tanh(c));
        }
        xf = xn;
        wave_sync();
    }
    __syncthreads();

    // ============== bottleneck: z = h@Wenc^T + b ; dec_in = z@Wdec^T + b ==============
    if (tid < RPB) {
        int row = tid, wv = row >> 4, rr = row & 15;
        float h[HENC];
#pragma unroll
        for (int k = 0; k < HENC; ++k) h[k] = (float)sH[wv][rr][k];
        float z[LATD];
#pragma unroll
        for (int l = 0; l < LATD; ++l) {
            float s = sBenc[l];
            for (int k = 0; k < HENC; ++k) s += h[k] * sWenc[l][k];
            z[l] = s;
            z_out[(size_t)(blockIdx.x * RPB + row) * LATD + l] = s;
        }
#pragma unroll
        for (int j = 0; j < HENC; ++j) {
            float s = sBdec[j];
#pragma unroll
            for (int l = 0; l < LATD; ++l) s += z[l] * sWdec[j][l];
            sX[wv][rr][j] = (_Float16)s;   // constant decoder input, K=32 fully used
        }
    }
    __syncthreads();
    for (int i = tid; i < NWAVE * RPW * 32; i += NTHR) ((_Float16*)sH)[i] = (_Float16)0.0f;
    __syncthreads();

    // decoder input-gate contribution + bias: computed ONCE, lives in registers
    v16h dfrag = load_a_frag(&sX[w][0][0], lane);
    v8f accInp[NTD];
#pragma unroll
    for (int jt = 0; jt < NTD; ++jt) {
        v8f a = {};
        v16h bw = *(const v16h*)&sWihD[jt][lane][0];
        a = __builtin_amdgcn_wmma_f32_16x16x32_f16(false, dfrag, false, bw,
                                                   (short)0, a, false, false);
        float bb = sBiasD[jt * 16 + nn];
#pragma unroll
        for (int r = 0; r < 8; ++r) a[r] += bb;
        accInp[jt] = a;
    }

    float cdL[8], cdH[8];
#pragma unroll
    for (int r = 0; r < 8; ++r) { cdL[r] = 0.0f; cdH[r] = 0.0f; }

    float* outBase = out + ((size_t)(bbase + mb) * T_SEQ) * HDEC + nn;

    // ================================ decoder ================================
    for (int t = 0; t < T_SEQ; ++t) {
        v16h hfrag = load_a_frag(&sH[w][0][0], lane);
        v8f acc[NTD];
#pragma unroll
        for (int jt = 0; jt < NTD; ++jt) {
            v16h bh = *(const v16h*)&sWhhD[jt][lane][0];
            acc[jt] = __builtin_amdgcn_wmma_f32_16x16x32_f16(false, hfrag, false, bh,
                                                             (short)0, accInp[jt], false, false);
        }
        // hid = nn (all lanes): bias already folded into accInp
#pragma unroll
        for (int r = 0; r < 8; ++r) {
            float gi = hw_sigmoid(acc[0][r]);
            float gf = hw_sigmoid(acc[2][r]);
            float gg = hw_tanh   (acc[4][r]);
            float go = hw_sigmoid(acc[6][r]);
            float c  = gf * cdL[r] + gi * gg;  cdL[r] = c;
            float hv = go * hw_tanh(c);
            sH[w][mb + r][nn] = (_Float16)hv;
            outBase[(size_t)r * T_SEQ * HDEC + (size_t)t * HDEC] = hv;
        }
        // hid = 16 lives in the odd tiles at nn==0 only
        if (nn == 0) {
#pragma unroll
            for (int r = 0; r < 8; ++r) {
                float gi = hw_sigmoid(acc[1][r]);
                float gf = hw_sigmoid(acc[3][r]);
                float gg = hw_tanh   (acc[5][r]);
                float go = hw_sigmoid(acc[7][r]);
                float c  = gf * cdH[r] + gi * gg;  cdH[r] = c;
                float hv = go * hw_tanh(c);
                sH[w][mb + r][16] = (_Float16)hv;
                outBase[(size_t)r * T_SEQ * HDEC + (size_t)t * HDEC + 16] = hv;
            }
        }
        wave_sync();
    }
}

extern "C" void kernel_launch(void* const* d_in, const int* in_sizes, int n_in,
                              void* d_out, int out_size, void* d_ws, size_t ws_size,
                              hipStream_t stream) {
    (void)in_sizes; (void)n_in; (void)out_size; (void)d_ws; (void)ws_size;
    const float* x       = (const float*)d_in[0];
    const float* Wih_e   = (const float*)d_in[1];
    const float* Whh_e   = (const float*)d_in[2];
    const float* bih_e   = (const float*)d_in[3];
    const float* bhh_e   = (const float*)d_in[4];
    const float* W_encfc = (const float*)d_in[5];
    const float* b_encfc = (const float*)d_in[6];
    const float* W_decfc = (const float*)d_in[7];
    const float* b_decfc = (const float*)d_in[8];
    const float* Wih_d   = (const float*)d_in[9];
    const float* Whh_d   = (const float*)d_in[10];
    const float* bih_d   = (const float*)d_in[11];
    const float* bhh_d   = (const float*)d_in[12];

    float* out = (float*)d_out;
    float* z   = out + (size_t)B_TOT * T_SEQ * DIN;

    dim3 grid(B_TOT / RPB);   // 256 blocks
    dim3 block(NTHR);         // 64 threads = 2 waves, 16 batch rows each
    lstm_ae_kernel<<<grid, block, 0, stream>>>(
        x, Wih_e, Whh_e, bih_e, bhh_e, W_encfc, b_encfc,
        W_decfc, b_decfc, Wih_d, Whh_d, bih_d, bhh_d, out, z);
}
